// NodeModelIn_35089882808862
// MI455X (gfx1250) — compile-verified, matrix operand order checked by reading, not run on hardware
//
#include <hip/hip_runtime.h>

// ---------------------------------------------------------------------------
// Problem constants (from reference)
// ---------------------------------------------------------------------------
#define N_NODES   100000
#define N_EDGES   3200000
#define EDGE_OUT  32
#define HID       128
#define NODE_OUT  64
#define DIN       97          // 3*EDGE_OUT + 1
#define LDS_S     136         // padded stride in halves (16B-aligned, bank-friendly)

typedef __attribute__((ext_vector_type(16))) _Float16 v16h;
typedef __attribute__((ext_vector_type(8)))  _Float16 v8h;
typedef __attribute__((ext_vector_type(8)))  float    v8f;

// Order-preserving map float -> u32 so integer atomicMax == float max.
__device__ __forceinline__ unsigned enc_f32(float f) {
    unsigned b = __float_as_uint(f);
    return (b & 0x80000000u) ? ~b : (b | 0x80000000u);
}
__device__ __forceinline__ float dec_f32(unsigned k) {
    unsigned b = (k & 0x80000000u) ? (k & 0x7FFFFFFFu) : ~k;
    return __uint_as_float(b);
}

// ---------------------------------------------------------------------------
// Kernel 1: zero-init workspace (run every call; harness poisons ws)
// ---------------------------------------------------------------------------
__global__ void init_ws_kernel(float* __restrict__ seg_sum,
                               unsigned* __restrict__ maxkey,
                               float* __restrict__ cnt) {
    int i = blockIdx.x * blockDim.x + threadIdx.x;
    if (i < N_NODES * EDGE_OUT) { seg_sum[i] = 0.0f; maxkey[i] = 0u; }
    if (i < N_NODES)            cnt[i] = 0.0f;
}

// ---------------------------------------------------------------------------
// Kernel 2: edge scatter — segment sum / max / count via L2 atomics
// One thread per (edge, feature): coalesced edge_attr loads.
// ---------------------------------------------------------------------------
__global__ void scatter_edges_kernel(const float* __restrict__ edge_attr,
                                     const int*   __restrict__ edge_row,   // edge_index[0]
                                     float* __restrict__ seg_sum,
                                     unsigned* __restrict__ maxkey,
                                     float* __restrict__ cnt) {
    int t = blockIdx.x * blockDim.x + threadIdx.x;   // < 102.4M, fits int
    if (t >= N_EDGES * EDGE_OUT) return;
    int e = t >> 5;
    int f = t & 31;
    int row = edge_row[e];
    float v = edge_attr[t];
    atomicAdd(&seg_sum[row * EDGE_OUT + f], v);
    atomicMax(&maxkey[row * EDGE_OUT + f], enc_f32(v));
    if (f == 0) atomicAdd(&cnt[row], 1.0f);
}

// ---------------------------------------------------------------------------
// Kernel 3: fused MLP with v_wmma_f32_16x16x32_f16.
// 256 threads = 8 waves; each wave owns a 16-node M-tile.
// ---------------------------------------------------------------------------
__global__ void __launch_bounds__(256) mlp_wmma_kernel(
    const float* __restrict__ seg_sum, const unsigned* __restrict__ maxkey,
    const float* __restrict__ cnt,
    const float* __restrict__ u, const int* __restrict__ batch,
    const float* __restrict__ W1, const float* __restrict__ b1,
    const float* __restrict__ W2, const float* __restrict__ b2,
    float* __restrict__ out)
{
    // LDS: W1^T [col 0..127][k 0..127], W2^T [col 0..63][k 0..127],
    // and one [16 x 128] h-tile per wave (reused for the GEMM1 output).
    __shared__ __align__(16) _Float16 w1t[HID      * LDS_S];
    __shared__ __align__(16) _Float16 w2t[NODE_OUT * LDS_S];
    __shared__ __align__(16) _Float16 htile[8 * 16 * LDS_S];

    const int tid = threadIdx.x;

    // Stage W1^T as f16 (K padded 97 -> 128 with zeros). Coalesced global reads.
    for (int idx = tid; idx < HID * HID; idx += 256) {
        int k   = idx >> 7;          // 0..127
        int col = idx & 127;         // fast-varying -> coalesced
        float v = (k < DIN) ? W1[k * HID + col] : 0.0f;
        w1t[col * LDS_S + k] = (_Float16)v;
    }
    // Stage W2^T as f16.
    for (int idx = tid; idx < HID * NODE_OUT; idx += 256) {
        int k   = idx >> 6;          // 0..127
        int col = idx & 63;
        w2t[col * LDS_S + k] = (_Float16)W2[k * NODE_OUT + col];
    }
    __syncthreads();

    const int wave = tid >> 5;
    const int lane = tid & 31;
    const int m0   = (blockIdx.x * 8 + wave) * 16;     // wave-uniform
    if (m0 >= N_NODES) return;                         // whole-wave exit; EXEC stays full

    _Float16* hw = &htile[wave * 16 * LDS_S];

    // ---- Stage h tile: [sum(32) | max(32) | mean(32) | u | 0-pad] as f16 ----
    for (int it = 0; it < 64; ++it) {
        int idx = lane + (it << 5);                    // 0..2047
        int r   = idx >> 7;                            // local row 0..15
        int col = idx & 127;
        int row = m0 + r; if (row >= N_NODES) row = N_NODES - 1;
        float cv = cnt[row];
        float v;
        if      (col < 32)  v = seg_sum[row * 32 + col];
        else if (col < 64)  v = (cv > 0.0f) ? dec_f32(maxkey[row * 32 + (col - 32)]) : 0.0f;
        else if (col < 96)  v = seg_sum[row * 32 + (col - 64)] / fmaxf(cv, 1.0f);
        else if (col == 96) v = u[batch[row]];
        else                v = 0.0f;
        hw[r * LDS_S + col] = (_Float16)v;
    }
    asm volatile("s_wait_dscnt 0" ::: "memory");       // cross-lane LDS RAW (same wave)

    const int rsel  = lane & 15;                       // row (A) / col (B/C) selector
    const int khalf = (lane >> 4) << 3;                // +8 K for upper half-wave
    const int rbase = (lane >> 4) << 3;                // C/D row base: 0 or 8

    // ---- Build A fragments for GEMM1 (ISA 16-bit A layout) ----
    v16h a1[4];
#pragma unroll
    for (int c = 0; c < 4; ++c) {
        int kc = c * 32 + khalf;
        v8h lo = *(const v8h*)&hw[rsel * LDS_S + kc];        // K = kc+0..7
        v8h hi = *(const v8h*)&hw[rsel * LDS_S + kc + 16];   // K = kc+16..23
#pragma unroll
        for (int e = 0; e < 8; ++e) { a1[c][e] = lo[e]; a1[c][8 + e] = hi[e]; }
    }

    // ---- GEMM1: [16 x 97] @ [97 x 128], bias + ReLU, back to LDS ----
#pragma unroll
    for (int n = 0; n < 8; ++n) {
        v8f acc = {};
        int col = n * 16 + rsel;
#pragma unroll
        for (int c = 0; c < 4; ++c) {
            int kc = c * 32 + khalf;
            v8h lo = *(const v8h*)&w1t[col * LDS_S + kc];
            v8h hi = *(const v8h*)&w1t[col * LDS_S + kc + 16];
            v16h b;
#pragma unroll
            for (int e = 0; e < 8; ++e) { b[e] = lo[e]; b[8 + e] = hi[e]; }
            acc = __builtin_amdgcn_wmma_f32_16x16x32_f16(
                false, a1[c], false, b, (short)0, acc, false, false);
        }
        float bias = b1[col];
#pragma unroll
        for (int r = 0; r < 8; ++r) {
            float v = acc[r] + bias;
            v = v > 0.0f ? v : 0.0f;                   // ReLU
            hw[(rbase + r) * LDS_S + col] = (_Float16)v;   // C-layout -> LDS reshuffle
        }
    }
    asm volatile("s_wait_dscnt 0" ::: "memory");

    // ---- Build A fragments for GEMM2 from the ReLU'd [16 x 128] tile ----
    v16h a2[4];
#pragma unroll
    for (int c = 0; c < 4; ++c) {
        int kc = c * 32 + khalf;
        v8h lo = *(const v8h*)&hw[rsel * LDS_S + kc];
        v8h hi = *(const v8h*)&hw[rsel * LDS_S + kc + 16];
#pragma unroll
        for (int e = 0; e < 8; ++e) { a2[c][e] = lo[e]; a2[c][8 + e] = hi[e]; }
    }

    // ---- GEMM2: [16 x 128] @ [128 x 64] + b2 -> out ----
#pragma unroll
    for (int n = 0; n < 4; ++n) {
        v8f acc = {};
        int col = n * 16 + rsel;
#pragma unroll
        for (int c = 0; c < 4; ++c) {
            int kc = c * 32 + khalf;
            v8h lo = *(const v8h*)&w2t[col * LDS_S + kc];
            v8h hi = *(const v8h*)&w2t[col * LDS_S + kc + 16];
            v16h b;
#pragma unroll
            for (int e = 0; e < 8; ++e) { b[e] = lo[e]; b[8 + e] = hi[e]; }
            acc = __builtin_amdgcn_wmma_f32_16x16x32_f16(
                false, a2[c], false, b, (short)0, acc, false, false);
        }
        float bias = b2[col];
#pragma unroll
        for (int r = 0; r < 8; ++r) {
            int row = m0 + rbase + r;
            if (row < N_NODES) out[row * NODE_OUT + col] = acc[r] + bias;
        }
    }
}

// ---------------------------------------------------------------------------
// Host launcher
// Inputs: 0:x 1:edge_index 2:edge_attr 3:u 4:batch 5:W1 6:b1 7:W2 8:b2
// ---------------------------------------------------------------------------
extern "C" void kernel_launch(void* const* d_in, const int* in_sizes, int n_in,
                              void* d_out, int out_size, void* d_ws, size_t ws_size,
                              hipStream_t stream) {
    const int*   edge_index = (const int*)  d_in[1];   // row = first N_EDGES entries
    const float* edge_attr  = (const float*)d_in[2];
    const float* u          = (const float*)d_in[3];
    const int*   batch      = (const int*)  d_in[4];
    const float* W1         = (const float*)d_in[5];
    const float* b1         = (const float*)d_in[6];
    const float* W2         = (const float*)d_in[7];
    const float* b2         = (const float*)d_in[8];
    float*       out        = (float*)d_out;

    char* ws = (char*)d_ws;
    float*    seg_sum = (float*)ws;                                        // 12.8 MB
    unsigned* maxkey  = (unsigned*)(ws + (size_t)N_NODES * EDGE_OUT * 4);  // 12.8 MB
    float*    cnt     = (float*)(ws + (size_t)N_NODES * EDGE_OUT * 8);     // 0.4 MB

    {
        int total = N_NODES * EDGE_OUT;
        int blocks = (total + 255) / 256;
        init_ws_kernel<<<blocks, 256, 0, stream>>>(seg_sum, maxkey, cnt);
    }
    {
        int total = N_EDGES * EDGE_OUT;                 // 102.4M threads
        int blocks = (total + 255) / 256;
        scatter_edges_kernel<<<blocks, 256, 0, stream>>>(edge_attr, edge_index,
                                                         seg_sum, maxkey, cnt);
    }
    {
        int blocks = (N_NODES + 127) / 128;             // 8 waves x 16 rows per block
        mlp_wmma_kernel<<<blocks, 256, 0, stream>>>(seg_sum, maxkey, cnt,
                                                    u, batch, W1, b1, W2, b2, out);
    }
}
